// HybridGNN_33423435498155
// MI455X (gfx1250) — matrix operand order repaired and checked by reading, not compiled, not run on hardware
//
#include <hip/hip_runtime.h>
#include <hip/hip_bf16.h>
#include <math.h>

typedef __attribute__((ext_vector_type(16))) _Float16 v16h;
typedef __attribute__((ext_vector_type(8)))  _Float16 v8h;
typedef __attribute__((ext_vector_type(8)))  float    v8f;

#define BN_EPS 1e-5f

// ---------------------------------------------------------------------------
// WMMA GEMM (f16 in, f32 out): C[M,N] = A[M,K] @ B[K,N] (+bias)(+ReLU)
// REQUIRES: M%16==0, K%32==0, N%32==0, A/B 16-byte aligned rows (K,N mult of 8).
// One wave computes a 16x32 output tile (two 16x16 WMMA tiles sharing the A
// fragment). Fragment layouts per CDNA5 ISA 7.12.2 (wave32):
//   A (16x32 f16): lane holds row M=lane&15; elems 0-7 -> K=kb+0..7,
//                  elems 8-15 -> K=kb+16..23, kb = (lane>=16)?8:0
//   B (32x16 f16): lane holds column N=lane&15; elems 0-7 -> K=kb+0..7,
//                  elems 8-15 -> K=kb+16..23 (same as A)
//   C/D (16x16 f32): vgpr j -> row j + ((lane>>4)<<3), col = lane&15
// A loads are unconditional global_load_b128 (8 halves each); B is read one half at a time.
// ---------------------------------------------------------------------------
__global__ void wmma_gemm_kernel(const _Float16* __restrict__ A, const _Float16* __restrict__ B,
                                 const float* __restrict__ bias, float* __restrict__ C,
                                 int M, int K, int N, int relu) {
  const int lane  = threadIdx.x & 31;
  const int tm    = blockIdx.x;                       // 16-row tile
  const int tn32  = blockIdx.y * blockDim.y + threadIdx.y;  // 32-col tile
  if (tn32 * 32 >= N) return;

  const int arow  = tm * 16 + (lane & 15);
  const int kbase = (lane >> 4) << 3;                 // 0 or 8
  const int cbase = tn32 * 32;

  const _Float16* aptr = A + (size_t)arow * K + kbase;
  const _Float16* bptr = B + (size_t)kbase * N + cbase + (lane & 15);
  const size_t bstep = (size_t)32 * N;                // 32 K-rows per iter

  v8f acc0 = {};
  v8f acc1 = {};
  for (int kk = 0; kk < K; kk += 32) {
    // A fragment: two contiguous 8-half runs
    const v8h a0 = *(const v8h*)(aptr);
    const v8h a1 = *(const v8h*)(aptr + 16);
    const v16h a = __builtin_shufflevector(a0, a1, 0,1,2,3,4,5,6,7,8,9,10,11,12,13,14,15);
    // B fragments for the two adjacent 16-col tiles (k order matches A)
    v16h b0, b1;
#pragma unroll
    for (int s = 0; s < 16; ++s) {
      const size_t off = (size_t)((s < 8) ? s : s + 8) * N;
      b0[s] = bptr[off];
      b1[s] = bptr[off + 16];
    }
    acc0 = __builtin_amdgcn_wmma_f32_16x16x32_f16(false, a, false, b0, (short)0, acc0, false, false);
    acc1 = __builtin_amdgcn_wmma_f32_16x16x32_f16(false, a, false, b1, (short)0, acc1, false, false);
    aptr += 32;
    bptr += bstep;
  }

  const int ccol  = cbase + (lane & 15);
  const int rbase = tm * 16 + ((lane >> 4) << 3);
  const float bv0 = bias ? bias[ccol]      : 0.0f;
  const float bv1 = bias ? bias[ccol + 16] : 0.0f;
#pragma unroll
  for (int j = 0; j < 8; ++j) {
    const int r = rbase + j;
    float v0 = acc0[j] + bv0;
    float v1 = acc1[j] + bv1;
    if (relu) { v0 = fmaxf(v0, 0.0f); v1 = fmaxf(v1, 0.0f); }
    C[(size_t)r * N + ccol]      = v0;
    C[(size_t)r * N + ccol + 16] = v1;
  }
}

// ---------------------------------------------------------------------------
// Conversion helpers
// ---------------------------------------------------------------------------
__global__ void f32_to_f16_kernel(const float* __restrict__ src, _Float16* __restrict__ dst, size_t n) {
  size_t i = (size_t)blockIdx.x * blockDim.x + threadIdx.x;
  size_t stride = (size_t)gridDim.x * blockDim.x;
  for (; i < n; i += stride) dst[i] = (_Float16)src[i];
}

// dst is [padRows x cols] f16; rows < srcRows converted from src, rest zero.
__global__ void f32_to_f16_padrows_kernel(const float* __restrict__ src, _Float16* __restrict__ dst,
                                          int srcRows, int cols, int padRows) {
  size_t i = (size_t)blockIdx.x * blockDim.x + threadIdx.x;
  size_t total = (size_t)padRows * cols;
  size_t stride = (size_t)gridDim.x * blockDim.x;
  for (; i < total; i += stride) {
    int r = (int)(i / cols);
    dst[i] = (r < srcRows) ? (_Float16)src[i] : (_Float16)0.0f;
  }
}

// ---------------------------------------------------------------------------
// Elementwise / reduction helpers
// ---------------------------------------------------------------------------
__global__ void fill_kernel(float* __restrict__ p, float v, size_t n) {
  size_t i = (size_t)blockIdx.x * blockDim.x + threadIdx.x;
  size_t stride = (size_t)gridDim.x * blockDim.x;
  for (; i < n; i += stride) p[i] = v;
}

__global__ void deg_count_kernel(const int* __restrict__ col, float* __restrict__ deg, int E) {
  int i = blockIdx.x * blockDim.x + threadIdx.x;
  int stride = gridDim.x * blockDim.x;
  for (; i < E; i += stride) atomicAdd(&deg[col[i]], 1.0f);
}

__global__ void rsqrt_inplace_kernel(float* __restrict__ p, int n) {
  int i = blockIdx.x * blockDim.x + threadIdx.x;
  if (i < n) p[i] = rsqrtf(p[i]);
}

// out[i,:] = xw[i,:] * dinv[i]^2   (self-loop term)
__global__ void self_loop_init_kernel(const float* __restrict__ xw, const float* __restrict__ dinv,
                                      float* __restrict__ out, int N, int F) {
  size_t i = (size_t)blockIdx.x * blockDim.x + threadIdx.x;
  size_t total = (size_t)N * F;
  size_t stride = (size_t)gridDim.x * blockDim.x;
  for (; i < total; i += stride) {
    int node = (int)(i / F);
    float d = dinv[node];
    out[i] = xw[i] * d * d;
  }
}

// wave-per-edge scatter: out[col[e],:] += xw[row[e],:] * dinv[row]*dinv[col]
__global__ void edge_aggregate_kernel(const float* __restrict__ xw, const int* __restrict__ rows,
                                      const int* __restrict__ cols, const float* __restrict__ dinv,
                                      float* __restrict__ out, int E, int F) {
  const int lane = threadIdx.x & 31;
  int wid = (int)(((size_t)blockIdx.x * blockDim.x + threadIdx.x) >> 5);
  const int nw = (gridDim.x * blockDim.x) >> 5;
  for (int e = wid; e < E; e += nw) {
    const int r = rows[e];
    const int c = cols[e];
    const float nrm = dinv[r] * dinv[c];
    const float* src = xw + (size_t)r * F;
    float* dst = out + (size_t)c * F;
    for (int f = lane; f < F; f += 32)
      atomicAdd(dst + f, src[f] * nrm);
  }
}

__global__ void bias_relu_kernel(float* __restrict__ x, const float* __restrict__ b, int M, int N) {
  size_t i = (size_t)blockIdx.x * blockDim.x + threadIdx.x;
  size_t total = (size_t)M * N;
  size_t stride = (size_t)gridDim.x * blockDim.x;
  for (; i < total; i += stride) {
    int col = (int)(i % N);
    x[i] = fmaxf(x[i] + b[col], 0.0f);
  }
}

// one thread per column per block; register accumulation, one atomic at end.
// launch: grid = (nblocks), block = N  (N <= 1024)
__global__ void bn_stats_kernel(const float* __restrict__ x, int M, int N,
                                float* __restrict__ sums, float* __restrict__ sumsq) {
  const int col = threadIdx.x;
  float s = 0.0f, s2 = 0.0f;
  for (int r = blockIdx.x; r < M; r += gridDim.x) {
    float v = x[(size_t)r * N + col];
    s += v;
    s2 += v * v;
  }
  atomicAdd(&sums[col], s);
  atomicAdd(&sumsq[col], s2);
}

__global__ void bn_apply_kernel(float* __restrict__ x, int M, int N,
                                const float* __restrict__ sums, const float* __restrict__ sumsq,
                                const float* __restrict__ gamma, const float* __restrict__ beta) {
  size_t i = (size_t)blockIdx.x * blockDim.x + threadIdx.x;
  size_t total = (size_t)M * N;
  size_t stride = (size_t)gridDim.x * blockDim.x;
  const float Minv = 1.0f / (float)M;
  for (; i < total; i += stride) {
    int col = (int)(i % N);
    float mu = sums[col] * Minv;
    float var = sumsq[col] * Minv - mu * mu;
    x[i] = gamma[col] * (x[i] - mu) * rsqrtf(var + BN_EPS) + beta[col];
  }
}

// wave-per-node pool accumulation into zbuf[:, 0:F] (leading dim ZLD)
__global__ void pool_accum_kernel(const float* __restrict__ h, const int* __restrict__ batch,
                                  float* __restrict__ zbuf, float* __restrict__ cnt,
                                  int N, int F, int ZLD) {
  const int lane = threadIdx.x & 31;
  int wid = (int)(((size_t)blockIdx.x * blockDim.x + threadIdx.x) >> 5);
  const int nw = (gridDim.x * blockDim.x) >> 5;
  for (int n = wid; n < N; n += nw) {
    const int g = batch[n];
    if (lane == 0) atomicAdd(&cnt[g], 1.0f);
    const float* src = h + (size_t)n * F;
    float* dst = zbuf + (size_t)g * ZLD;
    for (int f = lane; f < F; f += 32)
      atomicAdd(dst + f, src[f]);
  }
}

// cols 0..F-1: divide by count (mean); cols F..F+R-1: copy rdkit feats;
// cols >= F+R (zero padding) untouched.
__global__ void pool_finish_kernel(float* __restrict__ zbuf, const float* __restrict__ cnt,
                                   const float* __restrict__ rdkit, int G, int F, int R, int ZLD) {
  size_t i = (size_t)blockIdx.x * blockDim.x + threadIdx.x;
  size_t total = (size_t)G * ZLD;
  size_t stride = (size_t)gridDim.x * blockDim.x;
  for (; i < total; i += stride) {
    int g = (int)(i / ZLD);
    int col = (int)(i % ZLD);
    if (col < F) {
      zbuf[i] = zbuf[i] / fmaxf(cnt[g], 1.0f);
    } else if (col < F + R) {
      zbuf[i] = rdkit[(size_t)g * R + (col - F)];
    }
  }
}

// wave-per-row final projection: out[g] = dot(z2[g,:], w) + b
__global__ void final_dot_kernel(const float* __restrict__ z2, const float* __restrict__ w,
                                 const float* __restrict__ b, float* __restrict__ out,
                                 int G, int K) {
  const int lane = threadIdx.x & 31;
  int wid = (int)(((size_t)blockIdx.x * blockDim.x + threadIdx.x) >> 5);
  const int nw = (gridDim.x * blockDim.x) >> 5;
  for (int g = wid; g < G; g += nw) {
    const float* row = z2 + (size_t)g * K;
    float s = 0.0f;
    for (int k = lane; k < K; k += 32) s += row[k] * w[k];
#pragma unroll
    for (int off = 16; off > 0; off >>= 1) s += __shfl_down(s, off, 32);
    if (lane == 0) out[g] = s + b[0];
  }
}

// ---------------------------------------------------------------------------
// Host orchestration
// ---------------------------------------------------------------------------
static inline int cdiv(int a, int b) { return (a + b - 1) / b; }

static void launch_gemm(const _Float16* A, const _Float16* B, const float* bias, float* C,
                        int M, int K, int N, int relu, hipStream_t stream) {
  // one wave = 16x32 tile; 4 waves per block over adjacent n32 tiles
  dim3 grid(M / 16, cdiv(N / 32, 4));
  dim3 block(32, 4);
  wmma_gemm_kernel<<<grid, block, 0, stream>>>(A, B, bias, C, M, K, N, relu);
}

extern "C" void kernel_launch(void* const* d_in, const int* in_sizes, int n_in,
                              void* d_out, int out_size, void* d_ws, size_t ws_size,
                              hipStream_t stream) {
  // ---- inputs (setup_inputs dict order) ----
  const float* x      = (const float*)d_in[0];
  const int*   eidx   = (const int*)  d_in[1];
  const int*   batch  = (const int*)  d_in[2];
  const float* rdkit  = (const float*)d_in[3];
  const float* W1  = (const float*)d_in[4];
  const float* b1  = (const float*)d_in[5];
  const float* g1  = (const float*)d_in[6];
  const float* be1 = (const float*)d_in[7];
  const float* W2  = (const float*)d_in[8];
  const float* b2  = (const float*)d_in[9];
  const float* g2  = (const float*)d_in[10];
  const float* be2 = (const float*)d_in[11];
  const float* mW1  = (const float*)d_in[12];
  const float* mb1  = (const float*)d_in[13];
  const float* mg1  = (const float*)d_in[14];
  const float* mbe1 = (const float*)d_in[15];
  const float* mW2  = (const float*)d_in[16];
  const float* mb2  = (const float*)d_in[17];
  const float* mg2  = (const float*)d_in[18];
  const float* mbe2 = (const float*)d_in[19];
  const float* mW3  = (const float*)d_in[20];
  const float* mb3  = (const float*)d_in[21];
  float* out = (float*)d_out;

  // ---- dims ----
  const int E   = in_sizes[1] / 2;
  const int N   = in_sizes[2];            // 100000 nodes (mult of 16)
  const int F   = in_sizes[0] / N;        // 64
  const int H   = in_sizes[5];            // 128
  const int H2  = in_sizes[9];            // 256
  const int K1  = in_sizes[12] / H2;      // 456 = 2H + R
  const int R   = K1 - H2;                // 200
  const int G   = in_sizes[3] / R;        // 4096
  const int KP1 = ((K1 + 31) / 32) * 32;  // 480 (K padded for WMMA)
  const int ZLD = KP1;                    // padded concat row stride

  const int* row = eidx;                  // edge_index[0,:]
  const int* col = eidx + E;              // edge_index[1,:]

  // ---- workspace carve (256B aligned) ----
  char* p = (char*)d_ws;
  auto carve = [&](size_t bytes) -> void* {
    void* r = (void*)p;
    p += (bytes + 255) & ~(size_t)255;
    return r;
  };
  float*    dinv  = (float*)   carve((size_t)N * 4);           // degree -> rsqrt in place
  float*    xw1   = (float*)   carve((size_t)N * H  * 4);
  float*    h1    = (float*)   carve((size_t)N * H  * 4);
  float*    xw2   = (float*)   carve((size_t)N * H2 * 4);
  float*    h2    = (float*)   carve((size_t)N * H2 * 4);
  float*    sums  = (float*)   carve((size_t)H2 * 4);
  float*    sumsq = (float*)   carve((size_t)H2 * 4);
  float*    zbuf  = (float*)   carve((size_t)G * ZLD * 4);     // [emb | rdkit | 0pad]
  float*    cnt   = (float*)   carve((size_t)G * 4);
  float*    z1    = (float*)   carve((size_t)G * H2 * 4);
  float*    z2    = (float*)   carve((size_t)G * H  * 4);
  _Float16* xh    = (_Float16*)carve((size_t)N * F  * 2);      // f16 operand copies
  _Float16* W1h   = (_Float16*)carve((size_t)F * H  * 2);
  _Float16* h1h   = (_Float16*)carve((size_t)N * H  * 2);
  _Float16* W2h   = (_Float16*)carve((size_t)H * H2 * 2);
  _Float16* zh    = (_Float16*)carve((size_t)G * ZLD * 2);
  _Float16* mW1h  = (_Float16*)carve((size_t)KP1 * H2 * 2);
  _Float16* z1h   = (_Float16*)carve((size_t)G * H2 * 2);
  _Float16* mW2h  = (_Float16*)carve((size_t)H2 * H * 2);
  (void)ws_size; (void)n_in; (void)out_size;

  const dim3 blk256(256);
  const int egrid   = min(cdiv(E, 8), 65536 * 8);              // 8 waves/block
  const int ngridH  = cdiv(N * H, 256);
  const int ngridH2 = cdiv(N * H2, 256);

  // ===== degrees & symmetric normalization =====
  fill_kernel<<<cdiv(N, 256), blk256, 0, stream>>>(dinv, 1.0f, (size_t)N);
  deg_count_kernel<<<cdiv(E, 256), blk256, 0, stream>>>(col, dinv, E);
  rsqrt_inplace_kernel<<<cdiv(N, 256), blk256, 0, stream>>>(dinv, N);

  // ===== GCN layer 1: xw1 = x @ W1 ; aggregate ; +b1, ReLU ; BN =====
  f32_to_f16_kernel<<<cdiv(N * F, 256), blk256, 0, stream>>>(x, xh, (size_t)N * F);
  f32_to_f16_kernel<<<cdiv(F * H, 256), blk256, 0, stream>>>(W1, W1h, (size_t)F * H);
  launch_gemm(xh, W1h, nullptr, xw1, N, F, H, 0, stream);
  self_loop_init_kernel<<<ngridH, blk256, 0, stream>>>(xw1, dinv, h1, N, H);
  edge_aggregate_kernel<<<egrid, blk256, 0, stream>>>(xw1, row, col, dinv, h1, E, H);
  bias_relu_kernel<<<ngridH, blk256, 0, stream>>>(h1, b1, N, H);
  fill_kernel<<<1, blk256, 0, stream>>>(sums, 0.0f, (size_t)H);
  fill_kernel<<<1, blk256, 0, stream>>>(sumsq, 0.0f, (size_t)H);
  bn_stats_kernel<<<256, dim3(H), 0, stream>>>(h1, N, H, sums, sumsq);
  bn_apply_kernel<<<ngridH, blk256, 0, stream>>>(h1, N, H, sums, sumsq, g1, be1);

  // ===== GCN layer 2: xw2 = h1 @ W2 ; aggregate ; +b2, ReLU ; BN =====
  f32_to_f16_kernel<<<cdiv(N * H, 256), blk256, 0, stream>>>(h1, h1h, (size_t)N * H);
  f32_to_f16_kernel<<<cdiv(H * H2, 256), blk256, 0, stream>>>(W2, W2h, (size_t)H * H2);
  launch_gemm(h1h, W2h, nullptr, xw2, N, H, H2, 0, stream);
  self_loop_init_kernel<<<ngridH2, blk256, 0, stream>>>(xw2, dinv, h2, N, H2);
  edge_aggregate_kernel<<<egrid, blk256, 0, stream>>>(xw2, row, col, dinv, h2, E, H2);
  bias_relu_kernel<<<ngridH2, blk256, 0, stream>>>(h2, b2, N, H2);
  fill_kernel<<<1, blk256, 0, stream>>>(sums, 0.0f, (size_t)H2);
  fill_kernel<<<1, blk256, 0, stream>>>(sumsq, 0.0f, (size_t)H2);
  bn_stats_kernel<<<256, dim3(H2), 0, stream>>>(h2, N, H2, sums, sumsq);
  bn_apply_kernel<<<ngridH2, blk256, 0, stream>>>(h2, N, H2, sums, sumsq, g2, be2);

  // ===== global mean pool + concat rdkit -> zbuf [G, ZLD(padded)] =====
  fill_kernel<<<cdiv(G * ZLD, 256), blk256, 0, stream>>>(zbuf, 0.0f, (size_t)G * ZLD);
  fill_kernel<<<cdiv(G, 256), blk256, 0, stream>>>(cnt, 0.0f, (size_t)G);
  pool_accum_kernel<<<cdiv(N, 8), blk256, 0, stream>>>(h2, batch, zbuf, cnt, N, H2, ZLD);
  pool_finish_kernel<<<cdiv(G * ZLD, 256), blk256, 0, stream>>>(zbuf, cnt, rdkit, G, H2, R, ZLD);

  // ===== MLP layer 1: z1 = ReLU(zbuf @ mW1 + mb1) ; BN (K padded 456->480) =====
  f32_to_f16_kernel<<<cdiv(G * ZLD, 256), blk256, 0, stream>>>(zbuf, zh, (size_t)G * ZLD);
  f32_to_f16_padrows_kernel<<<cdiv(KP1 * H2, 256), blk256, 0, stream>>>(mW1, mW1h, K1, H2, KP1);
  launch_gemm(zh, mW1h, mb1, z1, G, KP1, H2, 1, stream);
  fill_kernel<<<1, blk256, 0, stream>>>(sums, 0.0f, (size_t)H2);
  fill_kernel<<<1, blk256, 0, stream>>>(sumsq, 0.0f, (size_t)H2);
  bn_stats_kernel<<<256, dim3(H2), 0, stream>>>(z1, G, H2, sums, sumsq);
  bn_apply_kernel<<<cdiv(G * H2, 256), blk256, 0, stream>>>(z1, G, H2, sums, sumsq, mg1, mbe1);

  // ===== MLP layer 2: z2 = ReLU(z1 @ mW2 + mb2) ; BN =====
  f32_to_f16_kernel<<<cdiv(G * H2, 256), blk256, 0, stream>>>(z1, z1h, (size_t)G * H2);
  f32_to_f16_kernel<<<cdiv(H2 * H, 256), blk256, 0, stream>>>(mW2, mW2h, (size_t)H2 * H);
  launch_gemm(z1h, mW2h, mb2, z2, G, H2, H, 1, stream);
  fill_kernel<<<1, blk256, 0, stream>>>(sums, 0.0f, (size_t)H);
  fill_kernel<<<1, blk256, 0, stream>>>(sumsq, 0.0f, (size_t)H);
  bn_stats_kernel<<<256, dim3(H), 0, stream>>>(z2, G, H, sums, sumsq);
  bn_apply_kernel<<<cdiv(G * H, 256), blk256, 0, stream>>>(z2, G, H, sums, sumsq, mg2, mbe2);

  // ===== final projection: out = z2 @ mW3 + mb3 =====
  final_dot_kernel<<<cdiv(G, 8), blk256, 0, stream>>>(z2, mW3, mb3, out, G, H);
}
